// OpenMaxClassifier_15118284882682
// MI455X (gfx1250) — compile-verified, matrix operand order checked by reading, not run on hardware
//
#include <hip/hip_runtime.h>
#include <hip/hip_bf16.h>

typedef __attribute__((ext_vector_type(16))) __bf16 v16bf;
typedef __attribute__((ext_vector_type(8)))  __bf16 v8bf;
typedef __attribute__((ext_vector_type(8)))  float  v8f;

#define Dk    256   // input dim (K of GEMM1)
#define H1n   256   // hidden1 (N of GEMM1, K of GEMM2)
#define H2n   128   // hidden2 (N of GEMM2, K of GEMM3)
#define NCLS  8
#define WAVES 8
#define ROWS_PER_WAVE 16
#define ROWS_PER_ITER (WAVES * ROWS_PER_WAVE)   // 128
#define ROW_ITERS 4
#define ROWS_PER_WG (ROWS_PER_ITER * ROW_ITERS) // 512

// ---- LDS layout (bytes) ----
// W1 fragments: 16 ntiles x 8 ktiles x 32 lanes x 32B = 131072  (nt-major!)
// W2 fragments: 8 x 8 x 32 x 32 = 65536
// [W3|V] fragments: 1 x 4 x 32 x 32 = 4096
#define W1F_OFF   0
#define W2F_OFF   131072
#define W3VF_OFF  (131072 + 65536)                 // 196608
#define B1_OFF    (W3VF_OFF + 4096)                // 200704
#define B2_OFF    (B1_OFF + 1024)                  // 201728
#define B3_OFF    (B2_OFF + 512)                   // 202240
#define AL_OFF    (B3_OFF + 32)
#define BE_OFF    (AL_OFF + 32)
#define TA_OFF    (BE_OFF + 32)
#define V2_OFF    (TA_OFF + 32)
#define STAGE_OFF 202752                           // 8 waves x 8192
#define ACC3_OFF  (STAGE_OFF + WAVES * 8192)       // 268288, 8 waves x 1024
#define SMEM_BYTES (ACC3_OFF + WAVES * 1024)       // 276480  (< 320KB WGP LDS)

extern __shared__ __align__(128) char smem[];

// Map matrix element (k, n) of a KxN bf16 B-operand into the pre-swizzled LDS
// fragment pool. Fragment order is nt-major so one N-tile's 8 fragments are a
// contiguous 8KiB window (all DS offsets fit the 16-bit immediate).
// Per CDNA5 16-bit operand layout:
//   lane = n%16 + 16*((k%32>>3)&1),  elem = (k&7) + 8*((k%32)>>4)
__device__ __forceinline__ int frag_byte(int k, int n, int kt_count) {
  int kt = k >> 5, kk = k & 31;
  int nt = n >> 4, nn = n & 15;
  int lane = nn + (((kk >> 3) & 1) << 4);
  int e = (kk & 7) + ((kk >> 4) << 3);
  return (((nt * kt_count + kt) * 32 + lane) << 5) + (e << 1);
}

// Load a 16-element A-fragment (M=lane&15 row, khalf=lane>>4) from fp32 memory:
// two contiguous 8-float runs at p[0..7] and p[16..23], converted to bf16.
__device__ __forceinline__ v16bf load_frag_f32(const float* p) {
  const float4* p4 = (const float4*)p;
  float4 q0 = p4[0], q1 = p4[1], q2 = p4[4], q3 = p4[5];
  v16bf t;
  t[0] = (__bf16)q0.x;  t[1] = (__bf16)q0.y;  t[2]  = (__bf16)q0.z;  t[3]  = (__bf16)q0.w;
  t[4] = (__bf16)q1.x;  t[5] = (__bf16)q1.y;  t[6]  = (__bf16)q1.z;  t[7]  = (__bf16)q1.w;
  t[8] = (__bf16)q2.x;  t[9] = (__bf16)q2.y;  t[10] = (__bf16)q2.z;  t[11] = (__bf16)q2.w;
  t[12] = (__bf16)q3.x; t[13] = (__bf16)q3.y; t[14] = (__bf16)q3.z;  t[15] = (__bf16)q3.w;
  return t;
}

__device__ __forceinline__ v16bf load_frag_bf16(const __bf16* p) {
  v8bf lo = *(const v8bf*)p;
  v8bf hi = *(const v8bf*)(p + 16);
  v16bf t;
#pragma unroll
  for (int e = 0; e < 8; ++e) { t[e] = lo[e]; t[e + 8] = hi[e]; }
  return t;
}

__global__ void openmax_kernel(const float* __restrict__ x,
                               const float* __restrict__ W1,
                               const float* __restrict__ b1,
                               const float* __restrict__ W2,
                               const float* __restrict__ b2,
                               const float* __restrict__ W3,
                               const float* __restrict__ b3,
                               const float* __restrict__ w_alpha,
                               const float* __restrict__ w_beta,
                               const float* __restrict__ w_tau,
                               const float* __restrict__ av,
                               float* __restrict__ out,
                               int Btotal) {
  const int tid = threadIdx.x;

  // ---------------- Stage weights into LDS, pre-swizzled to B-fragment order
  // W1: 256x256 fp32 -> bf16 fragments (KT=8)
  {
    const float4* w4 = (const float4*)W1;
#pragma unroll 4
    for (int it = 0; it < 64; ++it) {
      int flat4 = it * 256 + tid;              // float4 index, row has 64 float4
      int k = flat4 >> 6;
      int n0 = (flat4 & 63) << 2;
      float4 w = w4[flat4];
      float v[4] = {w.x, w.y, w.z, w.w};
#pragma unroll
      for (int j = 0; j < 4; ++j)
        *(__bf16*)(smem + W1F_OFF + frag_byte(k, n0 + j, 8)) = (__bf16)v[j];
    }
  }
  // W2: 256x128 fp32 -> bf16 fragments (KT=8)
  {
    const float4* w4 = (const float4*)W2;
#pragma unroll 4
    for (int it = 0; it < 32; ++it) {
      int flat4 = it * 256 + tid;              // row has 32 float4
      int k = flat4 >> 5;
      int n0 = (flat4 & 31) << 2;
      float4 w = w4[flat4];
      float v[4] = {w.x, w.y, w.z, w.w};
#pragma unroll
      for (int j = 0; j < 4; ++j)
        *(__bf16*)(smem + W2F_OFF + frag_byte(k, n0 + j, 8)) = (__bf16)v[j];
    }
  }
  // Combined B = [W3 (128x8) | activation_vectors^T (128x8)] -> 128x16, KT=4
#pragma unroll
  for (int it = 0; it < 8; ++it) {
    int flat = it * 256 + tid;
    int k = flat >> 4;
    int c = flat & 15;
    float v = (c < 8) ? W3[k * NCLS + c] : av[(c - 8) * H2n + k];
    *(__bf16*)(smem + W3VF_OFF + frag_byte(k, c, 4)) = (__bf16)v;
  }
  // Biases & Weibull params & ||v_c||^2
  ((float*)(smem + B1_OFF))[tid] = b1[tid];
  if (tid < 128) ((float*)(smem + B2_OFF))[tid] = b2[tid];
  if (tid < 8) {
    ((float*)(smem + B3_OFF))[tid] = b3[tid];
    ((float*)(smem + AL_OFF))[tid] = w_alpha[tid];
    ((float*)(smem + BE_OFF))[tid] = fmaxf(w_beta[tid], 1e-6f);
    ((float*)(smem + TA_OFF))[tid] = w_tau[tid];
    float s = 0.f;
    const float* vp = av + tid * H2n;
#pragma unroll 8
    for (int j = 0; j < H2n; ++j) s = fmaf(vp[j], vp[j], s);
    ((float*)(smem + V2_OFF))[tid] = s;
  }
  __syncthreads();

  const int wave = tid >> 5;
  const int lane = tid & 31;
  const int mrow = lane & 15;       // row within 16-row strip (and C column idx)
  const int khalf = lane >> 4;
  char* stage = smem + STAGE_OFF + wave * 8192;      // wave-private staging
  float* acc3buf = (float*)(smem + ACC3_OFF + wave * 1024);
  const float* b1s = (const float*)(smem + B1_OFF);
  const float* b2s = (const float*)(smem + B2_OFF);

  const size_t waveBase0 = (size_t)blockIdx.x * ROWS_PER_WG + (size_t)wave * ROWS_PER_WAVE;

  // ---- Preload A fragments of x for strip 0 (software pipeline prologue)
  v16bf a[8];
  {
    size_t rsrc = waveBase0 + mrow;
    if (rsrc >= (size_t)Btotal) rsrc = (size_t)Btotal - 1;
    const float* xrow = x + rsrc * Dk;
#pragma unroll
    for (int kt = 0; kt < 8; ++kt)
      a[kt] = load_frag_f32(xrow + kt * 32 + khalf * 8);
  }

  for (int iter = 0; iter < ROW_ITERS; ++iter) {
    size_t rowBase = waveBase0 + (size_t)iter * ROWS_PER_ITER;

    // ---- GEMM1: h = relu(x @ W1 + b1), staged to LDS as bf16 [16][256]
    for (int nt = 0; nt < 16; ++nt) {
      v16bf bfr[8];
#pragma unroll
      for (int kt = 0; kt < 8; ++kt)
        bfr[kt] = *(const v16bf*)(smem + W1F_OFF + (((nt * 8 + kt) * 32 + lane) << 5));
      v8f acc = {};
#pragma unroll
      for (int kt = 0; kt < 8; ++kt)
        acc = __builtin_amdgcn_wmma_f32_16x16x32_bf16(false, a[kt], false, bfr[kt],
                                                      (short)0, acc, false, false);
      float bias = b1s[nt * 16 + mrow];     // C-layout: column = lane&15
#pragma unroll
      for (int r = 0; r < 8; ++r) {
        float v = acc[r] + bias;
        v = v > 0.f ? v : 0.f;
        int m = r + (khalf << 3);
        *(__bf16*)(stage + (((m << 8) + nt * 16 + mrow) << 1)) = (__bf16)v;
      }
    }

    // ---- Prefetch next strip's x fragments into (now dead) a[]; overlaps
    //      with GEMM2/GEMM3/epilogue below to hide HBM latency.
    if (iter + 1 < ROW_ITERS) {
      size_t rsrc = rowBase + ROWS_PER_ITER + mrow;
      if (rsrc >= (size_t)Btotal) rsrc = (size_t)Btotal - 1;
      const float* xrow = x + rsrc * Dk;
#pragma unroll
      for (int kt = 0; kt < 8; ++kt)
        a[kt] = load_frag_f32(xrow + kt * 32 + khalf * 8);
    }

    // ---- A fragments of h from LDS staging
    v16bf hf[8];
#pragma unroll
    for (int kt = 0; kt < 8; ++kt)
      hf[kt] = load_frag_bf16((const __bf16*)stage + (mrow << 8) + kt * 32 + khalf * 8);

    // ---- GEMM2: act = relu(h @ W2 + b2), staged to LDS as fp32 [16][128]
    for (int nt = 0; nt < 8; ++nt) {
      v16bf bfr[8];
#pragma unroll
      for (int kt = 0; kt < 8; ++kt)
        bfr[kt] = *(const v16bf*)(smem + W2F_OFF + (((nt * 8 + kt) * 32 + lane) << 5));
      v8f acc = {};
#pragma unroll
      for (int kt = 0; kt < 8; ++kt)
        acc = __builtin_amdgcn_wmma_f32_16x16x32_bf16(false, hf[kt], false, bfr[kt],
                                                      (short)0, acc, false, false);
      float bias = b2s[nt * 16 + mrow];
#pragma unroll
      for (int r = 0; r < 8; ++r) {
        float v = acc[r] + bias;
        v = v > 0.f ? v : 0.f;
        int m = r + (khalf << 3);
        *(float*)(stage + (((m << 7) + nt * 16 + mrow) << 2)) = v;
      }
    }

    // ---- GEMM3: acc3 = act @ [W3 | V^T]  (16x16 result: cols 0-7 logits, 8-15 act.v)
    {
      v16bf bfr[4];
#pragma unroll
      for (int kt = 0; kt < 4; ++kt)
        bfr[kt] = *(const v16bf*)(smem + W3VF_OFF + ((kt * 32 + lane) << 5));
      v8f acc = {};
#pragma unroll
      for (int kt = 0; kt < 4; ++kt) {
        v16bf af = load_frag_f32((const float*)stage + (mrow << 7) + kt * 32 + khalf * 8);
        acc = __builtin_amdgcn_wmma_f32_16x16x32_bf16(false, af, false, bfr[kt],
                                                      (short)0, acc, false, false);
      }
#pragma unroll
      for (int r = 0; r < 8; ++r)
        acc3buf[((r + (khalf << 3)) << 4) + mrow] = acc[r];
    }

    // ---- OpenMax epilogue: one lane per row (lanes 0-15)
    if (lane < 16) {
      int m = lane;
      size_t row = rowBase + m;
      const float4* ar4 = (const float4*)((const float*)stage + (m << 7));
      float a2 = 0.f;
#pragma unroll
      for (int j = 0; j < 32; ++j) {
        float4 q = ar4[j];
        a2 = fmaf(q.x, q.x, a2); a2 = fmaf(q.y, q.y, a2);
        a2 = fmaf(q.z, q.z, a2); a2 = fmaf(q.w, q.w, a2);
      }
      const float* b3s = (const float*)(smem + B3_OFF);
      const float* als = (const float*)(smem + AL_OFF);
      const float* bes = (const float*)(smem + BE_OFF);
      const float* tas = (const float*)(smem + TA_OFF);
      const float* v2s = (const float*)(smem + V2_OFF);
      float logits[NCLS];
      float maxcdf = 0.f;
#pragma unroll
      for (int c = 0; c < NCLS; ++c) {
        float lg = acc3buf[(m << 4) + c] + b3s[c];
        float avd = acc3buf[(m << 4) + 8 + c];
        float d2 = a2 - 2.f * avd + v2s[c];
        float dist = sqrtf(fmaxf(d2, 0.f));
        float sx = fmaxf(dist - tas[c], 0.f);
        float cdf = 1.f - __expf(-__powf(sx / bes[c], als[c]));
        maxcdf = fmaxf(maxcdf, cdf);
        logits[c] = lg;
      }
      float scale = maxcdf > 0.5f ? 1.f - maxcdf : 1.f;
      if (row < (size_t)Btotal) {
        float* orow = out + row * NCLS;
        float4 o0 = {logits[0] * scale, logits[1] * scale, logits[2] * scale, logits[3] * scale};
        float4 o1 = {logits[4] * scale, logits[5] * scale, logits[6] * scale, logits[7] * scale};
        ((float4*)orow)[0] = o0;
        ((float4*)orow)[1] = o1;
      }
    }
  }
}

extern "C" void kernel_launch(void* const* d_in, const int* in_sizes, int n_in,
                              void* d_out, int out_size, void* d_ws, size_t ws_size,
                              hipStream_t stream) {
  const float* x  = (const float*)d_in[0];
  const float* W1 = (const float*)d_in[1];
  const float* b1 = (const float*)d_in[2];
  const float* W2 = (const float*)d_in[3];
  const float* b2 = (const float*)d_in[4];
  const float* W3 = (const float*)d_in[5];
  const float* b3 = (const float*)d_in[6];
  const float* al = (const float*)d_in[7];
  const float* be = (const float*)d_in[8];
  const float* ta = (const float*)d_in[9];
  const float* av = (const float*)d_in[10];
  float* out = (float*)d_out;

  int Btotal = in_sizes[0] / Dk;
  int grid = (Btotal + ROWS_PER_WG - 1) / ROWS_PER_WG;

  hipFuncSetAttribute((const void*)openmax_kernel,
                      hipFuncAttributeMaxDynamicSharedMemorySize, SMEM_BYTES);

  openmax_kernel<<<grid, 256, SMEM_BYTES, stream>>>(
      x, W1, b1, W2, b2, W3, b3, al, be, ta, av, out, Btotal);
}
